// LoFTREncoderLayer_54941221650801
// MI455X (gfx1250) — matrix operand order repaired and checked by reading, not compiled, not run on hardware
//
#include <hip/hip_runtime.h>
#include <hip/hip_bf16.h>
#include <math.h>

// ---------------------------------------------------------------------------
// LoFTR encoder layer (linear attention) for MI455X / gfx1250, wave32 + WMMA.
// Heavy GEMMs: 128x128-tile bf16 WMMA (f32 accum) with double-buffered LDS
// fed by CDNA5 async global->LDS copies (ASYNCcnt).
// ---------------------------------------------------------------------------

typedef unsigned short u16;
typedef __attribute__((ext_vector_type(16))) __bf16 v16bf;
typedef __attribute__((ext_vector_type(8)))  float  v8f;
typedef __attribute__((ext_vector_type(8)))  u16    u16x8;

union FragBF {
  v16bf v;
  u16x8 h[2];
};

__device__ __forceinline__ u16 f2bf(float f) {
  unsigned int u = __float_as_uint(f);
  u += 0x7FFFu + ((u >> 16) & 1u);   // round-to-nearest-even
  return (u16)(u >> 16);
}
__device__ __forceinline__ float bf2f(u16 h) {
  return __uint_as_float(((unsigned int)h) << 16);
}

// CDNA5 async global->LDS copy (16B per lane), tracked by ASYNCcnt.
__device__ __forceinline__ void async_copy_b128(unsigned int lds_off,
                                                const void* gptr) {
  asm volatile("global_load_async_to_lds_b128 %0, %1, off"
               :: "v"(lds_off), "v"(gptr) : "memory");
}
__device__ __forceinline__ void wait_async0() {
  asm volatile("s_wait_asynccnt 0x0" ::: "memory");
}
// Low 32 bits of a flat shared-aperture address == LDS byte offset (ISA 10.2).
__device__ __forceinline__ unsigned int lds_off32(const void* p) {
  return (unsigned int)(unsigned long long)p;
}

static const int BSZ = 8;
static const int LSEQ = 4800;
static const int DMODEL = 256;
static const int NHEAD = 8;
static const long long NTOK = (long long)BSZ * LSEQ;   // 38400

// ---------------------------------------------------------------------------
// WMMA GEMM: C[M,N] = A[M,K](bf16,row-major) x Bt[N,K](bf16, pre-transposed)
// Block: 256 threads = 8 waves, tile 128x128, wave tile 32x64 (2x4 wmma accs).
// Double-buffered LDS, async global->LDS staging overlapping WMMA.
// EPI 0: store f32 C ; EPI 1: store bf16 relu(C).
// Requires M%128==0, N%128==0, K%32==0 (true for all uses here).
// ---------------------------------------------------------------------------
#define GT_M 128
#define GT_N 128
#define GT_K 32

template <int EPI>
__global__ __launch_bounds__(256) void gemm_bf16_wmma(
    const u16* __restrict__ A, const u16* __restrict__ Bt,
    void* __restrict__ Cout, int M, int N, int K)
{
  __shared__ u16 As[2][GT_M * GT_K];
  __shared__ u16 Bs[2][GT_N * GT_K];

  const int bm   = blockIdx.x * GT_M;
  const int bn   = blockIdx.y * GT_N;
  const int tid  = threadIdx.x;
  const int wid  = tid >> 5;
  const int lane = tid & 31;
  const int hf   = lane >> 4;   // which 16-lane half
  const int r    = lane & 15;
  const int wm   = (wid & 3) * 32;   // wave M offset in tile
  const int wn   = (wid >> 2) * 64;  // wave N offset in tile

  v8f acc[2][4];
#pragma unroll
  for (int mt = 0; mt < 2; ++mt)
#pragma unroll
    for (int nt = 0; nt < 4; ++nt)
#pragma unroll
      for (int e = 0; e < 8; ++e) acc[mt][nt][e] = 0.0f;

  const int ldr = tid >> 1;        // 0..127 : row of tile this thread stages
  const int seg = (tid & 1) * 16;  // 0 or 16 : 16-element segment in K

  // Per-thread LDS destinations for the async staging (32B per matrix).
  unsigned int aoff[2], boff[2];
#pragma unroll
  for (int bfi = 0; bfi < 2; ++bfi) {
    aoff[bfi] = lds_off32(&As[bfi][ldr * GT_K + seg]);
    boff[bfi] = lds_off32(&Bs[bfi][ldr * GT_K + seg]);
  }
  const u16* ga0 = A  + (size_t)(bm + ldr) * K + seg;
  const u16* gb0 = Bt + (size_t)(bn + ldr) * K + seg;

  // Issue a K-tile (async, 4 x 16B per thread) into buffer `bfi`.
  auto issue_tile = [&](int k0, int bfi) {
    async_copy_b128(aoff[bfi],      ga0 + k0);
    async_copy_b128(aoff[bfi] + 16, ga0 + k0 + 8);
    async_copy_b128(boff[bfi],      gb0 + k0);
    async_copy_b128(boff[bfi] + 16, gb0 + k0 + 8);
  };

  issue_tile(0, 0);
  wait_async0();
  __syncthreads();

  const int nk = K / GT_K;
  for (int t = 0; t < nk; ++t) {
    const int cur = t & 1;
    if (t + 1 < nk) issue_tile((t + 1) * GT_K, cur ^ 1);

    // A fragment (16x32): lane r = row; elems 0..7 = K 8*hf.., 8..15 = 16+8*hf..
    FragBF fa[2];
#pragma unroll
    for (int mt = 0; mt < 2; ++mt) {
      const u16* base = &As[cur][(wm + mt * 16 + r) * GT_K];
      fa[mt].h[0] = *(const u16x8*)(base + 8 * hf);
      fa[mt].h[1] = *(const u16x8*)(base + 16 + 8 * hf);
    }
    // B fragment (32x16), Bs is [n][k]: lane r = col n; elems 0..7 = K 8*hf.., 8..15 = 16+8*hf..
    FragBF fb[4];
#pragma unroll
    for (int nt = 0; nt < 4; ++nt) {
      const u16* base = &Bs[cur][(wn + nt * 16 + r) * GT_K];
      fb[nt].h[0] = *(const u16x8*)(base + 8 * hf);
      fb[nt].h[1] = *(const u16x8*)(base + 16 + 8 * hf);
    }
#pragma unroll
    for (int mt = 0; mt < 2; ++mt)
#pragma unroll
      for (int nt = 0; nt < 4; ++nt)
        acc[mt][nt] = __builtin_amdgcn_wmma_f32_16x16x32_bf16(
            false, fa[mt].v, false, fb[nt].v, (short)0, acc[mt][nt], false, false);

    // Wave-local: make sure our prefetch landed; barrier (implies dscnt wait)
    // guarantees everyone is done reading `cur` before it is overwritten.
    wait_async0();
    __syncthreads();
  }

  // Epilogue: C VGPR j -> row (j + 8*hf), col r within 16x16 tile.
#pragma unroll
  for (int mt = 0; mt < 2; ++mt) {
#pragma unroll
    for (int nt = 0; nt < 4; ++nt) {
      const int col = bn + wn + nt * 16 + r;
#pragma unroll
      for (int j = 0; j < 8; ++j) {
        const int row = bm + wm + mt * 16 + 8 * hf + j;
        const float vv = acc[mt][nt][j];
        if (EPI == 0)
          ((float*)Cout)[(size_t)row * N + col] = vv;
        else
          ((u16*)Cout)[(size_t)row * N + col] = f2bf(fmaxf(vv, 0.0f));
      }
    }
  }
}

// ---------------------------------------------------------------------------
// f32 -> bf16 convert
// ---------------------------------------------------------------------------
__global__ void cvt_f32_bf16(const float* __restrict__ X, u16* __restrict__ Y,
                             long long n) {
  long long i = (long long)blockIdx.x * blockDim.x + threadIdx.x;
  if (i < n) Y[i] = f2bf(X[i]);
}

// Weight transpose + convert: W[K][N] f32 -> Wt[N][K] bf16
__global__ void wt_cvt(const float* __restrict__ W, u16* __restrict__ Wt,
                       int K, int N) {
  long long i = (long long)blockIdx.x * blockDim.x + threadIdx.x;
  if (i >= (long long)K * N) return;
  int n = (int)(i % N), k = (int)(i / N);
  Wt[(size_t)n * K + k] = f2bf(W[i]);
}

// ---------------------------------------------------------------------------
// Prepare: apply the block's permutation and phi(x)=elu(x)+1, emit
//   Qb  [b, l, 256]   bf16  (phi(q), GEMM-A layout)
//   Kt  [b, h, 32, S] bf16  (phi(k), transposed for WMMA-A of K^T V)
//   Vt  [b, h, 32, S] bf16  (v, transposed for WMMA-B [n][k] layout)
// perm: 0 ident, 1 p13 (0,1,3,2), 2 p21 (0,2,1,3), 3 p32 (0,3,2,1)
// ---------------------------------------------------------------------------
__global__ void prep_kernel(const float* __restrict__ q,
                            const float* __restrict__ k,
                            const float* __restrict__ v,
                            u16* __restrict__ Qb, u16* __restrict__ Kt,
                            u16* __restrict__ Vt, int L, int perm) {
  long long idx = (long long)blockIdx.x * blockDim.x + threadIdx.x;
  if (idx >= (long long)BSZ * L * 256) return;
  const int c  = (int)(idx & 255);          // c = h'*32 + i'
  long long bl = idx >> 8;
  const int lp = (int)(bl % L);
  const int b  = (int)(bl / L);
  const int hq = c >> 5, iq = c & 31;

  int l, cs;
  if (perm == 0) { l = lp; cs = c; }
  else if (perm == 1) { l = lp; cs = (c & 7) * 32 + (c >> 3); }
  else if (perm == 2) {
    long long f = (long long)lp * 256 + c;     // over [h, L, dh]
    int hi = (int)(f / ((long long)L * 32));
    long long rem = f % ((long long)L * 32);
    l = (int)(rem / 32);
    cs = hi * 32 + (int)(rem & 31);
  } else {
    long long f = (long long)lp * 256 + c;     // over [dh, h, L]
    int ii = (int)(f / ((long long)NHEAD * L));
    long long rem = f % ((long long)NHEAD * L);
    int hi = (int)(rem / L);
    l = (int)(rem % L);
    cs = hi * 32 + ii;
  }
  const size_t src = ((size_t)b * L + l) * 256 + cs;
  const float qv = q[src], kv = k[src], vv = v[src];
  const float phiq = qv > 0.0f ? qv + 1.0f : __expf(qv);  // elu(x)+1
  const float phik = kv > 0.0f ? kv + 1.0f : __expf(kv);

  Qb[idx] = f2bf(phiq);
  const size_t td = ((size_t)(b * NHEAD + hq) * 32 + iq) * (size_t)L + lp;
  Kt[td] = f2bf(phik);
  Vt[td] = f2bf(vv);
}

// ---------------------------------------------------------------------------
// Ksum[b,h,i] = sum_s Kt[b,h,i,s]   (one wave per row)
// ---------------------------------------------------------------------------
__global__ __launch_bounds__(256) void ksum_kernel(const u16* __restrict__ Kt,
                                                   float* __restrict__ Ksum,
                                                   int S) {
  const int row  = blockIdx.x * 8 + (threadIdx.x >> 5);  // bs*h*32 rows
  const int lane = threadIdx.x & 31;
  const u16* p = Kt + (size_t)row * S;
  float s = 0.0f;
  for (int i = lane; i < S; i += 32) s += bf2f(p[i]);
#pragma unroll
  for (int off = 16; off > 0; off >>= 1) s += __shfl_down(s, off, 32);
  if (lane == 0) Ksum[row] = s;
}

// ---------------------------------------------------------------------------
// KV per (b,h): C[32,32] = K^T[32,S] x V[S,32] via WMMA, split-S over 8 waves,
// reduced through LDS. Output transposed -> KVt[b,h][n][k] bf16 (GEMM-B layout).
// ---------------------------------------------------------------------------
__global__ __launch_bounds__(256) void kv_kernel(const u16* __restrict__ Kt,
                                                 const u16* __restrict__ Vt,
                                                 u16* __restrict__ KVt, int S) {
  const int bh   = blockIdx.x;  // b*NHEAD + h
  const u16* Kp  = Kt + (size_t)bh * 32 * S;
  const u16* Vp  = Vt + (size_t)bh * 32 * S;
  const int tid  = threadIdx.x;
  const int wid  = tid >> 5;
  const int lane = tid & 31;
  const int hf   = lane >> 4;
  const int r    = lane & 15;

  v8f acc[2][2];
#pragma unroll
  for (int mt = 0; mt < 2; ++mt)
#pragma unroll
    for (int nt = 0; nt < 2; ++nt)
#pragma unroll
      for (int e = 0; e < 8; ++e) acc[mt][nt][e] = 0.0f;

  const int nsteps = S / 32;  // 150
  for (int st = wid; st < nsteps; st += 8) {
    const int s0 = st * 32;
    FragBF fa[2], fb[2];
#pragma unroll
    for (int mt = 0; mt < 2; ++mt) {
      const u16* base = Kp + (size_t)(mt * 16 + r) * S + s0;
      fa[mt].h[0] = *(const u16x8*)(base + 8 * hf);
      fa[mt].h[1] = *(const u16x8*)(base + 16 + 8 * hf);
    }
#pragma unroll
    for (int nt = 0; nt < 2; ++nt) {
      const u16* base = Vp + (size_t)(nt * 16 + r) * S + s0;
      fb[nt].h[0] = *(const u16x8*)(base + 8 * hf);
      fb[nt].h[1] = *(const u16x8*)(base + 16 + 8 * hf);
    }
#pragma unroll
    for (int mt = 0; mt < 2; ++mt)
#pragma unroll
      for (int nt = 0; nt < 2; ++nt)
        acc[mt][nt] = __builtin_amdgcn_wmma_f32_16x16x32_bf16(
            false, fa[mt].v, false, fb[nt].v, (short)0, acc[mt][nt], false, false);
  }

  __shared__ float red[8][1024];  // 32 KB of 320 KB LDS
#pragma unroll
  for (int mt = 0; mt < 2; ++mt)
#pragma unroll
    for (int nt = 0; nt < 2; ++nt)
#pragma unroll
      for (int j = 0; j < 8; ++j)
        red[wid][(mt * 16 + 8 * hf + j) * 32 + (nt * 16 + r)] = acc[mt][nt][j];
  __syncthreads();
  for (int e = tid; e < 1024; e += 256) {
    float s = 0.0f;
#pragma unroll
    for (int w = 0; w < 8; ++w) s += red[w][e];
    const int m = e >> 5, n = e & 31;
    KVt[(size_t)bh * 1024 + n * 32 + m] = f2bf(s);  // transposed store
  }
}

// ---------------------------------------------------------------------------
// Z[b,l,h] = 1 / (phi(q)[b,l,h,:] . Ksum[b,h,:] + eps)
// ---------------------------------------------------------------------------
__global__ void z_kernel(const u16* __restrict__ Qb,
                         const float* __restrict__ Ksum,
                         float* __restrict__ Zf, int L) {
  long long idx = (long long)blockIdx.x * blockDim.x + threadIdx.x;
  if (idx >= (long long)BSZ * L * NHEAD) return;
  const int hh = (int)(idx & 7);
  long long bl = idx >> 3;
  const int b = (int)(bl / L);
  const u16* qp = Qb + (size_t)bl * 256 + hh * 32;
  const float* ks = Ksum + (size_t)(b * NHEAD + hh) * 32;
  float dot = 0.0f;
#pragma unroll
  for (int i = 0; i < 32; ++i) dot += bf2f(qp[i]) * ks[i];
  Zf[idx] = 1.0f / (dot + 1e-6f);
}

// ---------------------------------------------------------------------------
// msg[b,l,h,:] = (phi(q)[b,l,h,:] @ KV[b,h]) * Z[b,l,h]; one K=32 WMMA step.
// Grid: (ceil(L/128), bs*h). Wave handles 16 tokens x 32 outputs.
// ---------------------------------------------------------------------------
__global__ __launch_bounds__(256) void msg_kernel(const u16* __restrict__ Qb,
                                                  const u16* __restrict__ KVt,
                                                  const float* __restrict__ Zf,
                                                  u16* __restrict__ Msg, int L) {
  const int bh = blockIdx.y;
  const int b = bh >> 3, hh = bh & 7;
  const int tid  = threadIdx.x;
  const int wid  = tid >> 5;
  const int lane = tid & 31;
  const int hf   = lane >> 4;
  const int r    = lane & 15;
  const int tw = blockIdx.x * 128 + wid * 16;  // wave's first token
  if (tw >= L) return;  // wave-uniform (L % 16 == 0)

  FragBF fa;
  {
    const u16* base = Qb + ((size_t)b * L + tw + r) * 256 + hh * 32;
    fa.h[0] = *(const u16x8*)(base + 8 * hf);
    fa.h[1] = *(const u16x8*)(base + 16 + 8 * hf);
  }
  FragBF fb[2];
  const u16* bb = KVt + (size_t)bh * 1024;  // [n][k]
#pragma unroll
  for (int nt = 0; nt < 2; ++nt) {
    const u16* base = bb + (size_t)(nt * 16 + r) * 32;
    fb[nt].h[0] = *(const u16x8*)(base + 8 * hf);
    fb[nt].h[1] = *(const u16x8*)(base + 16 + 8 * hf);
  }
  v8f acc[2];
#pragma unroll
  for (int nt = 0; nt < 2; ++nt) {
#pragma unroll
    for (int e = 0; e < 8; ++e) acc[nt][e] = 0.0f;
    acc[nt] = __builtin_amdgcn_wmma_f32_16x16x32_bf16(
        false, fa.v, false, fb[nt].v, (short)0, acc[nt], false, false);
  }
#pragma unroll
  for (int j = 0; j < 8; ++j) {
    const int tok = tw + 8 * hf + j;
    const float z = Zf[((size_t)b * L + tok) * NHEAD + hh];
#pragma unroll
    for (int nt = 0; nt < 2; ++nt)
      Msg[((size_t)b * L + tok) * 256 + hh * 32 + nt * 16 + r] =
          f2bf(acc[nt][j] * z);
  }
}

// ---------------------------------------------------------------------------
// Fill left half of hcat (ld 512) with bf16(x_base).
// ---------------------------------------------------------------------------
__global__ void fill_left(const float* __restrict__ xb, u16* __restrict__ Hcat,
                          long long n) {
  long long i = (long long)blockIdx.x * blockDim.x + threadIdx.x;
  if (i >= n) return;
  const long long row = i >> 8;
  const int c = (int)(i & 255);
  Hcat[(size_t)row * 512 + c] = f2bf(xb[i]);
}

// ---------------------------------------------------------------------------
// LayerNorm over 256 channels; one wave per row.
// mode 0: write bf16 to Hcat right half (ld 512, +256)
// mode 1: write f32 to dst          (dst[row*256+c] = o)
// mode 2: accumulate f32 into dst   (dst[row*256+c] += o)
// ---------------------------------------------------------------------------
__global__ __launch_bounds__(256) void ln_kernel(const float* __restrict__ X,
                                                 const float* __restrict__ g,
                                                 const float* __restrict__ bt,
                                                 void* __restrict__ dst,
                                                 int mode) {
  const int row  = blockIdx.x * 8 + (threadIdx.x >> 5);
  const int lane = threadIdx.x & 31;
  const float* xr = X + (size_t)row * 256;
  float vals[8];
  float s = 0.0f;
#pragma unroll
  for (int j = 0; j < 8; ++j) { vals[j] = xr[j * 32 + lane]; s += vals[j]; }
#pragma unroll
  for (int off = 16; off > 0; off >>= 1) s += __shfl_down(s, off, 32);
  const float mean = __shfl(s, 0, 32) * (1.0f / 256.0f);
  float s2 = 0.0f;
#pragma unroll
  for (int j = 0; j < 8; ++j) { float d = vals[j] - mean; s2 += d * d; }
#pragma unroll
  for (int off = 16; off > 0; off >>= 1) s2 += __shfl_down(s2, off, 32);
  const float var = __shfl(s2, 0, 32) * (1.0f / 256.0f);
  const float rstd = rsqrtf(var + 1e-5f);
#pragma unroll
  for (int j = 0; j < 8; ++j) {
    const int c = j * 32 + lane;
    const float o = (vals[j] - mean) * rstd * g[c] + bt[c];
    if (mode == 0)
      ((u16*)dst)[(size_t)row * 512 + 256 + c] = f2bf(o);
    else if (mode == 1)
      ((float*)dst)[(size_t)row * 256 + c] = o;
    else
      ((float*)dst)[(size_t)row * 256 + c] += o;
  }
}

// out = base + 0.5 * macc
__global__ void combine_kernel(const float* __restrict__ base,
                               const float* __restrict__ macc,
                               float* __restrict__ out, long long n) {
  long long i = (long long)blockIdx.x * blockDim.x + threadIdx.x;
  if (i < n) out[i] = base[i] + 0.5f * macc[i];
}

// ---------------------------------------------------------------------------
// Host-side orchestration
// ---------------------------------------------------------------------------
extern "C" void kernel_launch(void* const* d_in, const int* in_sizes, int n_in,
                              void* d_out, int out_size, void* d_ws,
                              size_t ws_size, hipStream_t stream) {
  (void)in_sizes; (void)n_in; (void)out_size; (void)ws_size;
  const float* x   = (const float*)d_in[0];
  const float* src = (const float*)d_in[1];
  const float* Wq  = (const float*)d_in[2];
  const float* Wk  = (const float*)d_in[3];
  const float* Wv  = (const float*)d_in[4];
  const float* Wm  = (const float*)d_in[5];
  const float* W1  = (const float*)d_in[6];
  const float* W2  = (const float*)d_in[7];
  const float* g1  = (const float*)d_in[8];
  const float* b1  = (const float*)d_in[9];
  const float* g2  = (const float*)d_in[10];
  const float* b2  = (const float*)d_in[11];
  float* out = (float*)d_out;

  const long long NE = NTOK * DMODEL;  // 9,830,400 elements

  char* p = (char*)d_ws;
  auto alloc = [&](size_t bytes) -> void* {
    void* r = (void*)p;
    p += (bytes + 255) & ~(size_t)255;
    return r;
  };
  u16* xb   = (u16*)alloc(NE * 2);
  u16* srcb = (u16*)alloc(NE * 2);
  u16* Wqt  = (u16*)alloc(256 * 256 * 2);
  u16* Wkt  = (u16*)alloc(256 * 256 * 2);
  u16* Wvt  = (u16*)alloc(256 * 256 * 2);
  u16* Wmt  = (u16*)alloc(256 * 256 * 2);
  u16* W1t  = (u16*)alloc(512 * 512 * 2);
  u16* W2t  = (u16*)alloc(256 * 512 * 2);
  float* qf = (float*)alloc(NE * 4);
  float* kf = (float*)alloc(NE * 4);
  float* vf = (float*)alloc(NE * 4);
  u16* Qb   = (u16*)alloc(NE * 2);
  u16* Kt   = (u16*)alloc(NE * 2);
  u16* Vt   = (u16*)alloc(NE * 2);
  float* Ksum = (float*)alloc((size_t)BSZ * NHEAD * 32 * 4);
  u16* KVt  = (u16*)alloc((size_t)BSZ * NHEAD * 1024 * 2);
  float* Zf = (float*)alloc((size_t)BSZ * LSEQ * NHEAD * 4);
  u16* Msg  = (u16*)alloc(NE * 2);
  float* mw = (float*)alloc(NE * 4);
  u16* Hcat = (u16*)alloc(NTOK * 512 * 2);
  u16* Hrel = (u16*)alloc(NTOK * 512 * 2);
  float* Of = (float*)alloc(NE * 4);
  float* macc = (float*)alloc(NE * 4);
  float* xmid = (float*)alloc(NE * 4);

  const int T = 256;
  const int gNE = (int)((NE + T - 1) / T);

  auto gemmF = [&](const u16* A, const u16* Bt, float* C, int M, int N, int K) {
    dim3 g(M / GT_M, N / GT_N);
    gemm_bf16_wmma<0><<<g, T, 0, stream>>>(A, Bt, (void*)C, M, N, K);
  };
  auto gemmR = [&](const u16* A, const u16* Bt, u16* C, int M, int N, int K) {
    dim3 g(M / GT_M, N / GT_N);
    gemm_bf16_wmma<1><<<g, T, 0, stream>>>(A, Bt, (void*)C, M, N, K);
  };

  // --- conversions / weight transposes (once) ---
  cvt_f32_bf16<<<gNE, T, 0, stream>>>(x, xb, NE);
  cvt_f32_bf16<<<gNE, T, 0, stream>>>(src, srcb, NE);
  wt_cvt<<<(256 * 256 + T - 1) / T, T, 0, stream>>>(Wq, Wqt, 256, 256);
  wt_cvt<<<(256 * 256 + T - 1) / T, T, 0, stream>>>(Wk, Wkt, 256, 256);
  wt_cvt<<<(256 * 256 + T - 1) / T, T, 0, stream>>>(Wv, Wvt, 256, 256);
  wt_cvt<<<(256 * 256 + T - 1) / T, T, 0, stream>>>(Wm, Wmt, 256, 256);
  wt_cvt<<<(512 * 512 + T - 1) / T, T, 0, stream>>>(W1, W1t, 512, 512);
  wt_cvt<<<(512 * 256 + T - 1) / T, T, 0, stream>>>(W2, W2t, 512, 256);

  // --- projections ---
  gemmF(xb,   Wqt, qf, (int)NTOK, 256, 256);
  gemmF(srcb, Wkt, kf, (int)NTOK, 256, 256);
  gemmF(srcb, Wvt, vf, (int)NTOK, 256, 256);

  auto run_block = [&](int perm, const float* xbase, int accmode, bool refill) {
    prep_kernel<<<gNE, T, 0, stream>>>(qf, kf, vf, Qb, Kt, Vt, LSEQ, perm);
    ksum_kernel<<<(BSZ * NHEAD * 32) / 8, T, 0, stream>>>(Kt, Ksum, LSEQ);
    kv_kernel<<<BSZ * NHEAD, T, 0, stream>>>(Kt, Vt, KVt, LSEQ);
    z_kernel<<<(int)(((long long)BSZ * LSEQ * NHEAD + T - 1) / T), T, 0,
               stream>>>(Qb, Ksum, Zf, LSEQ);
    dim3 mg((LSEQ + 127) / 128, BSZ * NHEAD);
    msg_kernel<<<mg, T, 0, stream>>>(Qb, KVt, Zf, Msg, LSEQ);
    gemmF(Msg, Wmt, mw, (int)NTOK, 256, 256);
    if (refill) fill_left<<<gNE, T, 0, stream>>>(xbase, Hcat, NE);
    ln_kernel<<<(int)(NTOK / 8), T, 0, stream>>>(mw, g1, b1, (void*)Hcat, 0);
    gemmR(Hcat, W1t, Hrel, (int)NTOK, 512, 512);
    gemmF(Hrel, W2t, Of, (int)NTOK, 256, 512);
    ln_kernel<<<(int)(NTOK / 8), T, 0, stream>>>(Of, g2, b2, (void*)macc,
                                                 accmode);
  };

  // blocks 0,1 on x -> x_mid
  run_block(0, x, 1, true);
  run_block(1, x, 2, false);
  combine_kernel<<<gNE, T, 0, stream>>>(x, macc, xmid, NE);
  // blocks 2,3 on x_mid -> out
  run_block(2, xmid, 1, true);
  run_block(3, xmid, 2, false);
  combine_kernel<<<gNE, T, 0, stream>>>(xmid, macc, out, NE);
}
